// MultiHeadAttention_84164179132923
// MI455X (gfx1250) — matrix operand order repaired and checked by reading, not compiled, not run on hardware
//
#include <hip/hip_runtime.h>
#include <hip/hip_bf16.h>

// ---------------------------------------------------------------------------
// Shapes: BS=64, L=512, D=1024, H=16, DK=64.  M = BS*L = 32768.
// ---------------------------------------------------------------------------
#define BS_ 64
#define L_  512
#define D_  1024
#define H_  16
#define M_  (BS_ * L_)     // 32768
#define K2_ (2 * D_)       // 2048
#define K3_ (3 * D_)       // 3072

typedef __attribute__((ext_vector_type(16))) __bf16 bf16x16;
typedef __attribute__((ext_vector_type(8)))  float  f32x8;

union ABf  { bf16x16 v; unsigned short u[16]; };

// Native bf16 convert (lowered to v_cvt_pk_bf16_f32 on gfx1250 — verified)
__device__ __forceinline__ unsigned short f32_to_bf16_rne(float f) {
  union { __bf16 h; unsigned short s; } x;
  x.h = (__bf16)f;
  return x.s;
}

__device__ __forceinline__ void cvt4(unsigned short* d, float4 f) {
  d[0] = f32_to_bf16_rne(f.x);
  d[1] = f32_to_bf16_rne(f.y);
  d[2] = f32_to_bf16_rne(f.z);
  d[3] = f32_to_bf16_rne(f.w);
}

__device__ __forceinline__ f32x8 wmma_bf16(const ABf& a, const ABf& b, f32x8 c) {
  // D = A(16x32 bf16) x B(32x16 bf16) + C(16x16 f32)
  return __builtin_amdgcn_wmma_f32_16x16x32_bf16(false, a.v, false, b.v,
                                                 (short)0, c, false, false);
}

// ---------------------------------------------------------------------------
// CDNA5 async global->LDS DMA (ASYNCcnt path), guarded so the file still
// compiles (with synchronous staging) on toolchains lacking the builtins.
// Builtin signature (from hipcc diagnostic): (AS1 v4i*, AS3 v4i*, imm, imm).
// ---------------------------------------------------------------------------
#if defined(__has_builtin)
#if __has_builtin(__builtin_amdgcn_global_load_async_to_lds_b128) && \
    __has_builtin(__builtin_amdgcn_s_wait_asynccnt)
#define HAVE_ASYNC_LDS 1
#endif
#endif

#ifdef HAVE_ASYNC_LDS
typedef __attribute__((__vector_size__(4 * sizeof(int)))) int v4i_t;
typedef __attribute__((address_space(1))) v4i_t as1_v4i;
typedef __attribute__((address_space(3))) v4i_t as3_v4i;
__device__ __forceinline__ void async_copy16(const void* g, void* l) {
  // generic->AS1 is bit-identical; generic->AS3 truncates to LDS offset.
  __builtin_amdgcn_global_load_async_to_lds_b128(
      (as1_v4i*)(uintptr_t)g, (as3_v4i*)(uintptr_t)l, 0, 0);
}
#endif

// ---------------------------------------------------------------------------
// Kernel 0: f32 -> bf16 weight conversion (one-time pass over Wv, Wo)
// ---------------------------------------------------------------------------
__global__ void cvt_bf16_kernel(const float* __restrict__ src,
                                unsigned short* __restrict__ dst, int n) {
  int i = blockIdx.x * blockDim.x + threadIdx.x;
  if (i < n) dst[i] = f32_to_bf16_rne(src[i]);
}

// ---------------------------------------------------------------------------
// Kernel 1: qp[b][c] = q[b]·Wq[c] + bq[c]      (64 x 1024, K=2048) — tiny
// ---------------------------------------------------------------------------
__global__ void qp_kernel(const float* __restrict__ q,
                          const float* __restrict__ Wq,
                          const float* __restrict__ bq,
                          float* __restrict__ qp) {
  int i = blockIdx.x * blockDim.x + threadIdx.x;   // 65536
  int b = i >> 10, c = i & (D_ - 1);
  const float4* qv = (const float4*)(q + (size_t)b * K2_);
  const float4* wv = (const float4*)(Wq + (size_t)c * K2_);
  float s = 0.f;
  for (int e = 0; e < K2_ / 4; ++e) {
    float4 a = qv[e], w = wv[e];
    s += a.x * w.x + a.y * w.y + a.z * w.z + a.w * w.w;
  }
  qp[i] = s + bq[c];
}

// ---------------------------------------------------------------------------
// Kernel 2: u[b][h][e] = sum_d qp[b][h*64+d] * Wq[h*64+d][e]   — tiny
// ---------------------------------------------------------------------------
__global__ void u_kernel(const float* __restrict__ qp,
                         const float* __restrict__ Wq,
                         float* __restrict__ u) {
  int i = blockIdx.x * blockDim.x + threadIdx.x;   // 64*16*2048 = 2097152
  int e  = i & (K2_ - 1);
  int bh = i >> 11;
  int b = bh >> 4, h = bh & 15;
  const float* qph = qp + (size_t)b * D_ + h * 64;
  const float* w   = Wq + (size_t)(h * 64) * K2_ + e;
  float s = 0.f;
  #pragma unroll 8
  for (int d = 0; d < 64; ++d) s += qph[d] * w[(size_t)d * K2_];
  u[i] = s;
}

__global__ void c0_kernel(const float* __restrict__ qp,
                          const float* __restrict__ bq,
                          float* __restrict__ c0) {
  int i = blockIdx.x * blockDim.x + threadIdx.x;   // 1024 = 64*16
  if (i >= BS_ * H_) return;
  int b = i >> 4, h = i & 15;
  float s = 0.f;
  for (int d = 0; d < 64; ++d) s += qp[(size_t)b * D_ + h * 64 + d] * bq[h * 64 + d];
  c0[i] = s;
}

// ---------------------------------------------------------------------------
// Kernel 3: scores_t[b][h][l] = (k[b][l]·u[b][h] + c0[b][h]) / 8
//   WMMA GEMM per batch: A = k[b] (512 x 2048), B = u[b]^T (2048 x 16).
//   One wave per (b, 16-row tile).  2048 waves total.  k is streamed once.
// ---------------------------------------------------------------------------
__global__ __launch_bounds__(256) void scores_wmma_kernel(
    const float* __restrict__ k,        // [64][512][2048]
    const float* __restrict__ u,        // [64][16][2048]
    const float* __restrict__ c0,       // [64][16]
    float* __restrict__ scores_t) {     // [64][16][512]
  const int wave = (blockIdx.x * blockDim.x + threadIdx.x) >> 5;
  const int lane = threadIdx.x & 31;
  const int b    = wave >> 5;           // 32 row-tiles per batch
  const int mt   = wave & 31;
  const int half = lane >> 4;
  const int nl   = lane & 15;
  const int lBase = mt * 16;

  const float* Arow = k + ((size_t)b * L_ + lBase + nl) * K2_;   // A: row m = nl
  const float* Bcol = u + ((size_t)b * H_ + nl) * K2_;           // B: col n = nl

  f32x8 acc = {};
  for (int kk = 0; kk < K2_; kk += 32) {
    ABf a;
    {
      const float* p0 = Arow + kk + half * 16;       // K = half*16 + 0..7
      const float* p1 = Arow + kk + half * 16 + 8;   // K = half*16 + 8..15
      cvt4(a.u + 0,  ((const float4*)p0)[0]);
      cvt4(a.u + 4,  ((const float4*)p0)[1]);
      cvt4(a.u + 8,  ((const float4*)p1)[0]);
      cvt4(a.u + 12, ((const float4*)p1)[1]);
    }
    ABf bb;
    {
      const float* p = Bcol + kk + half * 16;        // K = half*16 + e
      cvt4(bb.u + 0,  ((const float4*)p)[0]);
      cvt4(bb.u + 4,  ((const float4*)p)[1]);
      cvt4(bb.u + 8,  ((const float4*)p)[2]);
      cvt4(bb.u + 12, ((const float4*)p)[3]);
    }
    acc = wmma_bf16(a, bb, acc);
  }
  const float c = c0[b * H_ + nl];
  #pragma unroll
  for (int r = 0; r < 8; ++r) {
    int row = lBase + half * 8 + r;                  // D: M = r + half*8, N = nl
    scores_t[((size_t)b * H_ + nl) * L_ + row] = (acc[r] + c) * 0.125f;
  }
}

// ---------------------------------------------------------------------------
// Kernel 4: qes[b][i][h] = sum_j qe[b][i][j] * scores_t[b][h][j]
//   Also emits for_correlation_weight[b][i] = qes[b][i][0].
// ---------------------------------------------------------------------------
__global__ __launch_bounds__(256) void qescore_wmma_kernel(
    const float* __restrict__ qe,        // [64][512][512]
    const float* __restrict__ scores_t,  // [64][16][512]
    float* __restrict__ qes,             // [64][512][16]
    float* __restrict__ fcw) {           // [64][512]
  const int wave = (blockIdx.x * blockDim.x + threadIdx.x) >> 5;
  const int lane = threadIdx.x & 31;
  const int b    = wave >> 5;
  const int mt   = wave & 31;
  const int half = lane >> 4;
  const int nl   = lane & 15;
  const int iBase = mt * 16;

  const float* Arow = qe + ((size_t)b * L_ + iBase + nl) * L_;
  const float* Bcol = scores_t + ((size_t)b * H_ + nl) * L_;

  f32x8 acc = {};
  for (int kk = 0; kk < L_; kk += 32) {
    ABf a;
    {
      const float* p0 = Arow + kk + half * 16;
      const float* p1 = Arow + kk + half * 16 + 8;
      cvt4(a.u + 0,  ((const float4*)p0)[0]);
      cvt4(a.u + 4,  ((const float4*)p0)[1]);
      cvt4(a.u + 8,  ((const float4*)p1)[0]);
      cvt4(a.u + 12, ((const float4*)p1)[1]);
    }
    ABf bb;
    {
      const float* p = Bcol + kk + half * 16;
      cvt4(bb.u + 0,  ((const float4*)p)[0]);
      cvt4(bb.u + 4,  ((const float4*)p)[1]);
      cvt4(bb.u + 8,  ((const float4*)p)[2]);
      cvt4(bb.u + 12, ((const float4*)p)[3]);
    }
    acc = wmma_bf16(a, bb, acc);
  }
  #pragma unroll
  for (int r = 0; r < 8; ++r) {
    int i = iBase + half * 8 + r;
    qes[((size_t)b * L_ + i) * H_ + nl] = acc[r];
    if (nl == 0) fcw[(size_t)b * L_ + i] = acc[r];   // head 0, q index 0
  }
}

// ---------------------------------------------------------------------------
// Tiled-GEMM compute helpers (one K-step: A frag + 4 B frags + 4 WMMAs)
// ---------------------------------------------------------------------------
#define BM 128
#define BN 64
#define BK 32

__device__ __forceinline__ void gemm_step_f32A(
    const float (*__restrict__ Asf)[BK], const unsigned short (*__restrict__ Bs)[BK],
    int wv, int half, int nl, f32x8 acc[4]) {
  ABf a;
  const int am = wv * 16 + nl;
  #pragma unroll
  for (int e = 0; e < 16; ++e) {
    int K = half * 16 + e;
    a.u[e] = f32_to_bf16_rne(Asf[am][K]);
  }
  #pragma unroll
  for (int j = 0; j < 4; ++j) {
    ABf bb;
    const int bn = j * 16 + nl;
    #pragma unroll
    for (int e = 0; e < 16; ++e) bb.u[e] = Bs[bn][half * 16 + e];
    acc[j] = wmma_bf16(a, bb, acc[j]);
  }
}

__device__ __forceinline__ void gemm_step_bf16A(
    const unsigned short (*__restrict__ As)[BK], const unsigned short (*__restrict__ Bs)[BK],
    int wv, int half, int nl, f32x8 acc[4]) {
  ABf a;
  const int am = wv * 16 + nl;
  #pragma unroll
  for (int e = 0; e < 16; ++e) {
    int K = half * 16 + e;
    a.u[e] = As[am][K];
  }
  #pragma unroll
  for (int j = 0; j < 4; ++j) {
    ABf bb;
    const int bn = j * 16 + nl;
    #pragma unroll
    for (int e = 0; e < 16; ++e) bb.u[e] = Bs[bn][half * 16 + e];
    acc[j] = wmma_bf16(a, bb, acc[j]);
  }
}

// ---------------------------------------------------------------------------
// Kernel 5: vg = sigmoid(lam_h*cw + (1-lam_h)*qes) * (v @ Wv.T + bv)  -> bf16
//   Tiled WMMA GEMM: M=32768, N=1024, K=3072.  Block tile 128x64, BK=32.
//   8 waves/block, each wave computes a 16x64 strip (4 accumulators).
//   Double-buffered async global->LDS staging (5 async ops/wave/stage;
//   in-order completion => s_wait_asynccnt 5 releases the previous stage).
// ---------------------------------------------------------------------------
__global__ __launch_bounds__(256) void vg_wmma_kernel(
    const float* __restrict__ v,             // [32768][3072]
    const unsigned short* __restrict__ Wvb,  // [1024][3072] bf16
    const float* __restrict__ bv,            // [1024]
    const float* __restrict__ cw,            // [32768]  (= [64][512])
    const float* __restrict__ qes,           // [32768][16]
    const float* __restrict__ lam,           // [16]
    unsigned short* __restrict__ vg) {       // [32768][1024] bf16
  const int t    = threadIdx.x;
  const int wv   = t >> 5;
  const int lane = t & 31;
  const int half = lane >> 4;
  const int nl   = lane & 15;
  const int blockRow = blockIdx.x * BM;
  const int blockCol = blockIdx.y * BN;

  f32x8 acc[4] = {};
  const int NSTEP = K3_ / BK;   // 96

#ifdef HAVE_ASYNC_LDS
  __shared__ __align__(16) float          Asf[2][BM][BK];  // 32 KB
  __shared__ __align__(16) unsigned short Bs[2][BN][BK];   //  8 KB

  // A: 128x32 f32 = 1024 b128; 4/thread.  B: 64x32 bf16 = 256 b128; 1/thread.
  const int a_r = 0, dummy = a_r; (void)dummy;
  auto issue = [&](int kk, int p) {
    #pragma unroll
    for (int i = 0; i < 4; ++i) {
      int id = t + 256 * i;
      int r = id >> 3, seg = id & 7;
      async_copy16(v + (size_t)(blockRow + r) * K3_ + kk + seg * 4,
                   &Asf[p][r][seg * 4]);
    }
    int r = t >> 2, seg = t & 3;
    async_copy16(Wvb + (size_t)(blockCol + r) * K3_ + kk + seg * 8,
                 &Bs[p][r][seg * 8]);
  };

  issue(0, 0);
  for (int step = 0; step < NSTEP; ++step) {
    const int p = step & 1;
    if (step + 1 < NSTEP) {
      issue((step + 1) * BK, p ^ 1);
      __builtin_amdgcn_s_wait_asynccnt(5);   // stage `step` landed
    } else {
      __builtin_amdgcn_s_wait_asynccnt(0);
    }
    __syncthreads();
    gemm_step_f32A(Asf[p], Bs[p], wv, half, nl, acc);
    __syncthreads();
  }
#else
  __shared__ __align__(16) float          Asf[BM][BK];
  __shared__ __align__(16) unsigned short Bs[BN][BK];

  for (int step = 0; step < NSTEP; ++step) {
    const int kk = step * BK;
    { // A stage (f32, 16 floats per thread)
      int r = t >> 1, ah = (t & 1) * 16;
      const float4* ap = (const float4*)(v + (size_t)(blockRow + r) * K3_ + kk + ah);
      float4* dp = (float4*)&Asf[r][ah];
      dp[0] = ap[0]; dp[1] = ap[1]; dp[2] = ap[2]; dp[3] = ap[3];
    }
    { // B stage (bf16, 16 bytes per thread)
      int r = t >> 2, seg = (t & 3) * 8;
      *(uint4*)&Bs[r][seg] =
          *(const uint4*)(Wvb + (size_t)(blockCol + r) * K3_ + kk + seg);
    }
    __syncthreads();
    gemm_step_f32A(Asf, Bs, wv, half, nl, acc);
    __syncthreads();
  }
#endif

  // epilogue: gating + bias, store bf16
  const int h = blockCol >> 6;               // one head per 64-col block
  const float l0 = lam[h];
  float g[8];
  #pragma unroll
  for (int r = 0; r < 8; ++r) {
    int row = blockRow + wv * 16 + half * 8 + r;
    float x = l0 * cw[row] + (1.f - l0) * qes[(size_t)row * H_ + h];
    g[r] = 1.f / (1.f + expf(-x));
  }
  #pragma unroll
  for (int j = 0; j < 4; ++j) {
    int col = blockCol + j * 16 + nl;
    float bias = bv[col];
    #pragma unroll
    for (int r = 0; r < 8; ++r) {
      int row = blockRow + wv * 16 + half * 8 + r;
      vg[(size_t)row * D_ + col] = f32_to_bf16_rne((acc[j][r] + bias) * g[r]);
    }
  }
}

// ---------------------------------------------------------------------------
// Kernel 6: out = vg @ Wo.T + bo   (M=32768, N=1024, K=1024, bf16 x bf16)
//   Double-buffered async staging (3 async ops/wave/stage).
// ---------------------------------------------------------------------------
__global__ __launch_bounds__(256) void out_wmma_kernel(
    const unsigned short* __restrict__ vg,   // [32768][1024] bf16
    const unsigned short* __restrict__ Wob,  // [1024][1024]  bf16
    const float* __restrict__ bo,            // [1024]
    float* __restrict__ out) {               // [32768][1024]
  const int t    = threadIdx.x;
  const int wv   = t >> 5;
  const int lane = t & 31;
  const int half = lane >> 4;
  const int nl   = lane & 15;
  const int blockRow = blockIdx.x * BM;
  const int blockCol = blockIdx.y * BN;

  f32x8 acc[4] = {};
  const int NSTEP = D_ / BK;   // 32

#ifdef HAVE_ASYNC_LDS
  __shared__ __align__(16) unsigned short As[2][BM][BK];  // 16 KB
  __shared__ __align__(16) unsigned short Bs[2][BN][BK];  //  8 KB

  auto issue = [&](int kk, int p) {
    #pragma unroll
    for (int i = 0; i < 2; ++i) {
      int id = t + 256 * i;
      int r = id >> 2, seg = id & 3;
      async_copy16(vg + (size_t)(blockRow + r) * D_ + kk + seg * 8,
                   &As[p][r][seg * 8]);
    }
    int r = t >> 2, seg = t & 3;
    async_copy16(Wob + (size_t)(blockCol + r) * D_ + kk + seg * 8,
                 &Bs[p][r][seg * 8]);
  };

  issue(0, 0);
  for (int step = 0; step < NSTEP; ++step) {
    const int p = step & 1;
    if (step + 1 < NSTEP) {
      issue((step + 1) * BK, p ^ 1);
      __builtin_amdgcn_s_wait_asynccnt(3);   // stage `step` landed
    } else {
      __builtin_amdgcn_s_wait_asynccnt(0);
    }
    __syncthreads();
    gemm_step_bf16A(As[p], Bs[p], wv, half, nl, acc);
    __syncthreads();
  }
#else
  __shared__ __align__(16) unsigned short As[BM][BK];
  __shared__ __align__(16) unsigned short Bs[BN][BK];

  for (int step = 0; step < NSTEP; ++step) {
    const int kk = step * BK;
    { // A stage (bf16, 32B per thread)
      int r = t >> 1, ah = (t & 1) * 16;
      const uint4* ap = (const uint4*)(vg + (size_t)(blockRow + r) * D_ + kk + ah);
      *(uint4*)&As[r][ah]     = ap[0];
      *(uint4*)&As[r][ah + 8] = ap[1];
    }
    { // B stage
      int r = t >> 2, seg = (t & 3) * 8;
      *(uint4*)&Bs[r][seg] =
          *(const uint4*)(Wob + (size_t)(blockCol + r) * D_ + kk + seg);
    }
    __syncthreads();
    gemm_step_bf16A(As, Bs, wv, half, nl, acc);
    __syncthreads();
  }
#endif

  #pragma unroll
  for (int j = 0; j < 4; ++j) {
    int col = blockCol + j * 16 + nl;
    float bias = bo[col];
    #pragma unroll
    for (int r = 0; r < 8; ++r) {
      int row = blockRow + wv * 16 + half * 8 + r;
      out[(size_t)row * D_ + col] = acc[j][r] + bias;
    }
  }
}

// ---------------------------------------------------------------------------
// Host launcher
// ---------------------------------------------------------------------------
extern "C" void kernel_launch(void* const* d_in, const int* in_sizes, int n_in,
                              void* d_out, int out_size, void* d_ws, size_t ws_size,
                              hipStream_t stream) {
  (void)in_sizes; (void)n_in; (void)out_size; (void)ws_size;
  const float* q   = (const float*)d_in[0];   // [64][1][2048]
  const float* k   = (const float*)d_in[1];   // [64][512][2048]
  const float* v   = (const float*)d_in[2];   // [64][512][3072]
  const float* cw  = (const float*)d_in[3];   // [64][512]
  const float* qe  = (const float*)d_in[4];   // [64][512][512]
  const float* Wq  = (const float*)d_in[5];   // [1024][2048]
  const float* bq  = (const float*)d_in[6];   // [1024]
  const float* Wv  = (const float*)d_in[7];   // [1024][3072]
  const float* bv  = (const float*)d_in[8];   // [1024]
  const float* Wo  = (const float*)d_in[9];   // [1024][1024]
  const float* bo  = (const float*)d_in[10];  // [1024]
  const float* lam = (const float*)d_in[11];  // [16]

  float* out = (float*)d_out;                 // [32768][1024]
  float* fcw = out + (size_t)M_ * D_;         // [64][512]

  // workspace carve-up
  char* w = (char*)d_ws;
  unsigned short* Wvb = (unsigned short*)w; w += (size_t)D_ * K3_ * 2;   // 6 MB
  unsigned short* Wob = (unsigned short*)w; w += (size_t)D_ * D_  * 2;   // 2 MB
  float* qp       = (float*)w; w += (size_t)BS_ * D_ * 4;                // 256 KB
  float* u        = (float*)w; w += (size_t)BS_ * H_ * K2_ * 4;          // 8 MB
  float* c0       = (float*)w; w += (size_t)BS_ * H_ * 4;                // 4 KB
  float* scores_t = (float*)w; w += (size_t)BS_ * H_ * L_ * 4;           // 2 MB
  float* qes      = (float*)w; w += (size_t)BS_ * L_ * H_ * 4;           // 2 MB
  unsigned short* vg = (unsigned short*)w;                               // 64 MB

  // 0) weight conversions
  {
    int n = D_ * K3_;
    cvt_bf16_kernel<<<(n + 255) / 256, 256, 0, stream>>>(Wv, Wvb, n);
    n = D_ * D_;
    cvt_bf16_kernel<<<(n + 255) / 256, 256, 0, stream>>>(Wo, Wob, n);
  }
  // 1) qp
  qp_kernel<<<(BS_ * D_) / 256, 256, 0, stream>>>(q, Wq, bq, qp);
  // 2) u, c0
  u_kernel<<<(BS_ * H_ * K2_) / 256, 256, 0, stream>>>(qp, Wq, u);
  c0_kernel<<<4, 256, 0, stream>>>(qp, bq, c0);
  // 3) scores  (2048 waves = 256 blocks x 8 waves)
  scores_wmma_kernel<<<256, 256, 0, stream>>>(k, u, c0, scores_t);
  // 4) qe_score + for_correlation_weight
  qescore_wmma_kernel<<<256, 256, 0, stream>>>(qe, scores_t, qes, fcw);
  // 5) gated value projection -> vg (bf16)
  {
    dim3 grid(M_ / BM, D_ / BN);  // (256, 16)
    vg_wmma_kernel<<<grid, 256, 0, stream>>>(v, Wvb, bv, cw, qes, lam, vg);
  }
  // 6) output projection
  {
    dim3 grid(M_ / BM, D_ / BN);  // (256, 16)
    out_wmma_kernel<<<grid, 256, 0, stream>>>(vg, Wob, bo, out);
  }
}
